// SimplifiedMamba2Block_21827023798473
// MI455X (gfx1250) — matrix operand order repaired and checked by reading, not compiled, not run on hardware
//
#include <hip/hip_runtime.h>
#include <hip/hip_fp16.h>
#include <math.h>

typedef _Float16 v16h __attribute__((ext_vector_type(16)));
typedef _Float16 v8h  __attribute__((ext_vector_type(8)));
typedef float    v8f  __attribute__((ext_vector_type(8)));

#define D_MODEL 1024
#define D_STATE 8
#define D_INNER 1536
#define SEQLEN  2048
#define NBATCH  4
#define NTOK    (NBATCH * SEQLEN)   // 8192
#define LN_EPS  1e-5f

__device__ __forceinline__ float clip5(float v) {
    return fminf(fmaxf(v, -5.0f), 5.0f);
}

// ---------------------------------------------------------------------------
// f32 -> f16 weight conversion
// ---------------------------------------------------------------------------
__global__ __launch_bounds__(256) void cvt_f32_to_f16(const float* __restrict__ in,
                                                      _Float16* __restrict__ out, int n) {
    int i = blockIdx.x * 256 + threadIdx.x;
    if (i < n) out[i] = (_Float16)in[i];
}

// ---------------------------------------------------------------------------
// Input LayerNorm: one block per token; writes f16 normalized row
// ---------------------------------------------------------------------------
__global__ __launch_bounds__(256) void ln_in_kernel(const float* __restrict__ x,
                                                    const float* __restrict__ g,
                                                    const float* __restrict__ b,
                                                    _Float16* __restrict__ out) {
    __shared__ float red[256];
    const int t = blockIdx.x;
    const int tid = threadIdx.x;
    const float* row = x + (size_t)t * D_MODEL;

    float v[4];
    float s = 0.0f;
#pragma unroll
    for (int j = 0; j < 4; ++j) { v[j] = row[tid + j * 256]; s += v[j]; }
    red[tid] = s;
    __syncthreads();
    for (int off = 128; off > 0; off >>= 1) {
        if (tid < off) red[tid] += red[tid + off];
        __syncthreads();
    }
    const float mu = red[0] * (1.0f / D_MODEL);
    __syncthreads();

    s = 0.0f;
#pragma unroll
    for (int j = 0; j < 4; ++j) { float d = v[j] - mu; s += d * d; }
    red[tid] = s;
    __syncthreads();
    for (int off = 128; off > 0; off >>= 1) {
        if (tid < off) red[tid] += red[tid + off];
        __syncthreads();
    }
    const float rstd = rsqrtf(red[0] * (1.0f / D_MODEL) + LN_EPS);

    _Float16* orow = out + (size_t)t * D_MODEL;
#pragma unroll
    for (int j = 0; j < 4; ++j) {
        int c = tid + j * 256;
        orow[c] = (_Float16)((v[j] - mu) * rstd * g[c] + b[c]);
    }
}

// ---------------------------------------------------------------------------
// Output LayerNorm: in-place on d_out (row fully read before written)
// ---------------------------------------------------------------------------
__global__ __launch_bounds__(256) void ln_out_kernel(float* __restrict__ y,
                                                     const float* __restrict__ g,
                                                     const float* __restrict__ b) {
    __shared__ float red[256];
    const int t = blockIdx.x;
    const int tid = threadIdx.x;
    float* row = y + (size_t)t * D_MODEL;

    float v[4];
    float s = 0.0f;
#pragma unroll
    for (int j = 0; j < 4; ++j) { v[j] = row[tid + j * 256]; s += v[j]; }
    red[tid] = s;
    __syncthreads();
    for (int off = 128; off > 0; off >>= 1) {
        if (tid < off) red[tid] += red[tid + off];
        __syncthreads();
    }
    const float mu = red[0] * (1.0f / D_MODEL);
    __syncthreads();

    s = 0.0f;
#pragma unroll
    for (int j = 0; j < 4; ++j) { float d = v[j] - mu; s += d * d; }
    red[tid] = s;
    __syncthreads();
    for (int off = 128; off > 0; off >>= 1) {
        if (tid < off) red[tid] += red[tid + off];
        __syncthreads();
    }
    const float rstd = rsqrtf(red[0] * (1.0f / D_MODEL) + LN_EPS);
    __syncthreads();   // all reads done before in-place writes

#pragma unroll
    for (int j = 0; j < 4; ++j) {
        int c = tid + j * 256;
        row[c] = (v[j] - mu) * rstd * g[c] + b[c];
    }
}

// ---------------------------------------------------------------------------
// Async fill of one 128x32-half tile pair (A rows, B rows) into LDS.
// 512 16-byte chunks per tile; 256 threads x 2 chunks each, per matrix.
// Uses CDNA5 GLOBAL_LOAD_ASYNC_TO_LDS_B128 (ASYNCcnt-tracked).
// ---------------------------------------------------------------------------
__device__ __forceinline__ void async_fill_tile(const _Float16* __restrict__ gA,
                                                const _Float16* __restrict__ gB,
                                                int K, int rowA0, int rowB0, int k0,
                                                _Float16* sA, _Float16* sB, int tid) {
#pragma unroll
    for (int c = 0; c < 2; ++c) {
        const int id  = tid * 2 + c;      // 0..511
        const int row = id >> 2;          // 0..127
        const int seg = (id & 3) * 8;     // half offset within 32-half row

        const unsigned long long ga =
            (unsigned long long)(size_t)(gA + (size_t)(rowA0 + row) * K + k0 + seg);
        const unsigned la = (unsigned)(size_t)(sA + row * 32 + seg);
        asm volatile("global_load_async_to_lds_b128 %0, %1, off"
                     :: "v"(la), "v"(ga) : "memory");

        const unsigned long long gb =
            (unsigned long long)(size_t)(gB + (size_t)(rowB0 + row) * K + k0 + seg);
        const unsigned lb = (unsigned)(size_t)(sB + row * 32 + seg);
        asm volatile("global_load_async_to_lds_b128 %0, %1, off"
                     :: "v"(lb), "v"(gb) : "memory");
    }
}

// ---------------------------------------------------------------------------
// WMMA GEMM: C(MxN) = A(MxK,f16,row-major) * Bm(NxK,f16,row-major)^T
// Block = 256 threads = 8 waves; block tile 128x128; wave tile 64x32.
// Double-buffered async global->LDS staging, WMMA from LDS fragments.
// EPI==0: GEMM1 epilogue -> clip; cols [0,1536) to xssm f32,
//         cols [1536,3072) -> tanh -> tgate f16.
// EPI==1: GEMM2 epilogue -> clip -> out f32 (stride D_MODEL).
// ---------------------------------------------------------------------------
template <int EPI>
__global__ __launch_bounds__(256) void wmma_gemm_kernel(const _Float16* __restrict__ A,
                                                        const _Float16* __restrict__ Bm,
                                                        int K,
                                                        float* __restrict__ out_f32,
                                                        _Float16* __restrict__ out_f16) {
    __shared__ __align__(16) _Float16 sA[2][128 * 32];   // 8 KB per buffer
    __shared__ __align__(16) _Float16 sB[2][128 * 32];   // 8 KB per buffer

    const int tid    = threadIdx.x;
    const int lane   = tid & 31;
    const int wid    = tid >> 5;
    const int laneHi = lane >> 4;      // 0 | 1
    const int lmod   = lane & 15;
    const int waveM  = wid & 1;        // 2 waves in M
    const int waveN  = wid >> 1;       // 4 waves in N
    const int rowA0  = blockIdx.y * 128;
    const int rowB0  = blockIdx.x * 128;
    const int m0     = rowA0 + waveM * 64;
    const int n0     = rowB0 + waveN * 32;

    v8f acc[4][2];
#pragma unroll
    for (int mi = 0; mi < 4; ++mi)
#pragma unroll
        for (int ni = 0; ni < 2; ++ni) acc[mi][ni] = (v8f)(0.0f);

    // prologue: stage tile 0 (4 async issues per thread)
    async_fill_tile(A, Bm, K, rowA0, rowB0, 0, sA[0], sB[0], tid);

    for (int k0 = 0; k0 < K; k0 += 32) {
        const int cur = (k0 >> 5) & 1;

        if (k0 + 32 < K) {
            // prefetch next tile into other buffer, then wait for current tile
            async_fill_tile(A, Bm, K, rowA0, rowB0, k0 + 32, sA[cur ^ 1], sB[cur ^ 1], tid);
            asm volatile("s_wait_asynccnt 0x4" ::: "memory");
        } else {
            asm volatile("s_wait_asynccnt 0x0" ::: "memory");
        }
        __syncthreads();   // current tile visible to all waves

        v16h a[4], b[2];
        // A fragment: lanes 0-15 get K = {0..7, 16..23}; lanes 16-31 +8
#pragma unroll
        for (int mi = 0; mi < 4; ++mi) {
            const _Float16* p = &sA[cur][(waveM * 64 + mi * 16 + lmod) * 32 + laneHi * 8];
            v8h lo = *(const v8h*)p;
            v8h hi = *(const v8h*)(p + 16);
            a[mi] = __builtin_shufflevector(lo, hi, 0, 1, 2, 3, 4, 5, 6, 7,
                                            8, 9, 10, 11, 12, 13, 14, 15);
        }
        // B fragment: lane holds column n; lanes 0-15 get K = {0..7, 16..23}; lanes 16-31 +8
#pragma unroll
        for (int ni = 0; ni < 2; ++ni) {
            const _Float16* p = &sB[cur][(waveN * 32 + ni * 16 + lmod) * 32 + laneHi * 8];
            v8h lo = *(const v8h*)p;
            v8h hi = *(const v8h*)(p + 16);
            b[ni] = __builtin_shufflevector(lo, hi, 0, 1, 2, 3, 4, 5, 6, 7,
                                            8, 9, 10, 11, 12, 13, 14, 15);
        }
#pragma unroll
        for (int mi = 0; mi < 4; ++mi)
#pragma unroll
            for (int ni = 0; ni < 2; ++ni)
                acc[mi][ni] = __builtin_amdgcn_wmma_f32_16x16x32_f16(
                    false, a[mi], false, b[ni], (short)0, acc[mi][ni], false, false);

        __syncthreads();   // all waves done with `cur` before it is refilled
    }

    // Epilogue: D layout -> VGPR v = row (m + v + 8*laneHi), lane%16 = col
#pragma unroll
    for (int mi = 0; mi < 4; ++mi) {
#pragma unroll
        for (int ni = 0; ni < 2; ++ni) {
            const int col = n0 + ni * 16 + lmod;
#pragma unroll
            for (int v = 0; v < 8; ++v) {
                const int row = m0 + mi * 16 + v + laneHi * 8;
                const float val = clip5(acc[mi][ni][v]);
                if (EPI == 0) {
                    if (col < D_INNER)
                        out_f32[(size_t)row * D_INNER + col] = val;
                    else
                        out_f16[(size_t)row * D_INNER + (col - D_INNER)] =
                            (_Float16)tanhf(val);
                } else {
                    out_f32[(size_t)row * D_MODEL + col] = val;
                }
            }
        }
    }
}

// ---------------------------------------------------------------------------
// SSM recurrent scan: one thread per (batch, channel); lanes map to contiguous
// channels -> coalesced per-timestep loads/stores.
// ---------------------------------------------------------------------------
__global__ __launch_bounds__(256) void ssm_scan_kernel(const float* __restrict__ xssm,
                                                       const _Float16* __restrict__ tgate,
                                                       const float* __restrict__ A_log,
                                                       const float* __restrict__ Bp,
                                                       const float* __restrict__ Cp,
                                                       _Float16* __restrict__ g16out) {
    const int ch = blockIdx.x * 256 + threadIdx.x;     // 0 .. 4*1536-1
    if (ch >= NBATCH * D_INNER) return;
    const int bb = ch / D_INNER;
    const int i  = ch - bb * D_INNER;

    float dec[D_STATE], Bv[D_STATE], Cv[D_STATE], h[D_STATE];
#pragma unroll
    for (int d = 0; d < D_STATE; ++d) {
        float al = A_log[i * D_STATE + d];
        al = fminf(fmaxf(al, -5.0f), 0.0f);
        float Aa = -__expf(al);
        Aa = fminf(fmaxf(Aa, -2.0f), -0.01f);
        dec[d] = Aa * 0.9f;
        Bv[d] = Bp[i * D_STATE + d] * 0.1f;
        Cv[d] = Cp[i * D_STATE + d];
        h[d] = 0.0f;
    }

    size_t idx = (size_t)bb * SEQLEN * D_INNER + i;
    for (int t = 0; t < SEQLEN; ++t) {
        const float xv = xssm[idx];
        const float gv = (float)tgate[idx];
        float y = 0.0f;
#pragma unroll
        for (int d = 0; d < D_STATE; ++d) {
            h[d] = clip5(h[d] * dec[d] + xv * Bv[d]);
            y += h[d] * Cv[d];
        }
        y = clip5(y);
        g16out[idx] = (_Float16)(y * gv);
        idx += D_INNER;
    }
}

// ---------------------------------------------------------------------------
// Launch
// ---------------------------------------------------------------------------
extern "C" void kernel_launch(void* const* d_in, const int* in_sizes, int n_in,
                              void* d_out, int out_size, void* d_ws, size_t ws_size,
                              hipStream_t stream) {
    const float* x       = (const float*)d_in[0];
    const float* W_in    = (const float*)d_in[1];
    const float* W_out   = (const float*)d_in[2];
    const float* A_log   = (const float*)d_in[3];
    const float* Bp      = (const float*)d_in[4];
    const float* Cp      = (const float*)d_in[5];
    const float* ln_in_g = (const float*)d_in[6];
    const float* ln_in_b = (const float*)d_in[7];
    const float* ln_out_g= (const float*)d_in[8];
    const float* ln_out_b= (const float*)d_in[9];
    float* out = (float*)d_out;

    // workspace layout (256B aligned offsets)
    char* ws = (char*)d_ws;
    _Float16* A16    = (_Float16*)(ws);                               // 8192*1024 f16 = 16 MB
    _Float16* Win16  = (_Float16*)(ws + (size_t)16777216);            // 3072*1024 f16 =  6 MB
    _Float16* Wout16 = (_Float16*)(ws + (size_t)23068672);            // 1024*1536 f16 =  3 MB
    float*    xssm   = (float*)   (ws + (size_t)26214400);            // 8192*1536 f32 = 48 MB
    _Float16* tgate  = (_Float16*)(ws + (size_t)76546048);            // 8192*1536 f16 = 24 MB
    _Float16* G16    = (_Float16*)(ws + (size_t)101711872);           // 8192*1536 f16 = 24 MB

    // 1) weight conversion to f16
    {
        int n1 = 2 * D_INNER * D_MODEL;   // 3,145,728
        cvt_f32_to_f16<<<(n1 + 255) / 256, 256, 0, stream>>>(W_in, Win16, n1);
        int n2 = D_MODEL * D_INNER;       // 1,572,864
        cvt_f32_to_f16<<<(n2 + 255) / 256, 256, 0, stream>>>(W_out, Wout16, n2);
    }

    // 2) input layernorm -> f16
    ln_in_kernel<<<NTOK, 256, 0, stream>>>(x, ln_in_g, ln_in_b, A16);

    // 3) GEMM1: (8192 x 1024) x (1024 x 3072) -> xssm f32 / tanh(gate) f16
    {
        dim3 grid(2 * D_INNER / 128, NTOK / 128);   // (24, 64)
        wmma_gemm_kernel<0><<<grid, 256, 0, stream>>>(A16, Win16, D_MODEL, xssm, tgate);
    }

    // 4) SSM scan + gate multiply -> G16 f16
    ssm_scan_kernel<<<(NBATCH * D_INNER) / 256, 256, 0, stream>>>(
        xssm, tgate, A_log, Bp, Cp, G16);

    // 5) GEMM2: (8192 x 1536) x (1536 x 1024) -> d_out f32 (pre-LN, clipped)
    {
        dim3 grid(D_MODEL / 128, NTOK / 128);       // (8, 64)
        wmma_gemm_kernel<1><<<grid, 256, 0, stream>>>(G16, Wout16, D_INNER, out, nullptr);
    }

    // 6) output layernorm in place on d_out
    ln_out_kernel<<<NTOK, 256, 0, stream>>>(out, ln_out_g, ln_out_b);
}
